// StreamTensorBase_85658827751771
// MI455X (gfx1250) — matrix operand order repaired and checked by reading, not compiled, not run on hardware
//
#include <hip/hip_runtime.h>
#include <hip/hip_bf16.h>

typedef _Float16 v16h __attribute__((ext_vector_type(16)));
typedef _Float16 v8h  __attribute__((ext_vector_type(8)));
typedef float    v8f  __attribute__((ext_vector_type(8)));

#define GRD   128
#define G2    (GRD*GRD)
#define G3    (GRD*GRD*GRD)
#define NRAYS 4096
#define NSAMP 256
#define NPTS  (NRAYS*NSAMP)
#define APPC  27
#define FCW   128          // hidden width (== K pad)
#define INDIM 102
#define ROWS  128          // points per block
#define THREADS 256

__device__ __forceinline__ float softplus_f(float x) {
    return (x > 20.f) ? x : log1pf(__expf(x));
}
__device__ __forceinline__ float sigmoid_f(float x) {
    return 1.f / (1.f + __expf(-x));
}

// One wave computes a 16-row x 128-col strip of D = A(16x128) * B(128x128),
// accumulating into acc[8] (8 n-tiles of 16 cols). A is an LDS row-major f16
// tile [ROWS][FCW]; Bt is the weight matrix TRANSPOSED in LDS: Bt[n][k].
// Fragment layouts used here (wave32; A and B share a contiguous k order):
//   A 16x32 f16 : lanes 0-15 row M=lane, K = k0+{0..15}
//                 lanes 16-31 row M=lane-16, K = k0+{16..31}
//   B 32x16 f16 : lanes 0-15 col N=lane, K = k0+{0..15}
//                 lanes 16-31 col N=lane-16, K = k0+{16..31}
//   C/D 16x16 f32: VGPR r -> row r (lanes 0-15) / row r+8 (lanes 16-31), col = lane&15
__device__ __forceinline__ void wave_gemm_16x128(const _Float16 (*A)[FCW],
                                                 const _Float16 (*Bt)[FCW],
                                                 int wv, int lane, v8f acc[8])
{
    const int arow = wv * 16 + (lane & 15);
    const int kAlo = (lane < 16) ? 0 : 16;
    const int kAhi = (lane < 16) ? 8 : 24;
    const int ncol = lane & 15;
    const int kB   = (lane < 16) ? 0 : 16;

#pragma unroll
    for (int kt = 0; kt < 4; ++kt) {
        const int k0 = kt * 32;
        v8h alo = *(const v8h*)&A[arow][k0 + kAlo];
        v8h ahi = *(const v8h*)&A[arow][k0 + kAhi];
        v16h a;
#pragma unroll
        for (int j = 0; j < 8; ++j) { a[j] = alo[j]; a[j + 8] = ahi[j]; }
#pragma unroll
        for (int nt = 0; nt < 8; ++nt) {
            v16h b = *(const v16h*)&Bt[nt * 16 + ncol][k0 + kB];
            acc[nt] = __builtin_amdgcn_wmma_f32_16x16x32_f16(
                false, a, false, b, (short)0, acc[nt], false, false);
        }
    }
}

__global__ __launch_bounds__(THREADS, 1)
void nerf_feat_mlp_kernel(const float* __restrict__ xyz,
                          const float* __restrict__ viewdirs,
                          const float* __restrict__ density_vol,
                          const float* __restrict__ app_vol,
                          const float* __restrict__ W1, const float* __restrict__ b1,
                          const float* __restrict__ W2, const float* __restrict__ b2,
                          const float* __restrict__ W3, const float* __restrict__ b3,
                          float* __restrict__ sigma_out,
                          float* __restrict__ rgb_out)
{
    __shared__ _Float16 Alds[ROWS][FCW];   // 32 KB: mlp_in tile (f16, K padded to 128)
    __shared__ _Float16 W1t[FCW][FCW];     // 32 KB: W1^T  (Bt[n][k])
    __shared__ _Float16 W2t[FCW][FCW];     // 32 KB: W2^T
    __shared__ _Float16 Hlds[ROWS][FCW];   // 32 KB: hidden activations
    __shared__ float b1s[FCW], b2s[FCW];
    __shared__ float W3s[FCW * 3];
    __shared__ float b3s[3];

    const int tid = threadIdx.x;
    const int blk = blockIdx.x;

    if (tid >= ROWS) {
        // ---- staging waves: weights -> LDS (transposed, f32 -> f16) ----
        const int t = tid - ROWS;                       // 0..127
        for (int i = t; i < FCW * FCW; i += ROWS) {
            const int n = i & (FCW - 1);
            const int k = i >> 7;
            W1t[n][k] = (k < INDIM) ? (_Float16)W1[k * FCW + n] : (_Float16)0.f;
            W2t[n][k] = (_Float16)W2[k * FCW + n];
        }
        if (t < FCW) { b1s[t] = b1[t]; b2s[t] = b2[t]; }
        for (int i = t; i < FCW * 3; i += ROWS) W3s[i] = W3[i];
        if (t < 3) b3s[t] = b3[t];
    } else {
        // ---- feature waves: one point per thread ----
        const int p = blk * ROWS + tid;
        const float px = xyz[3 * p + 0], py = xyz[3 * p + 1], pz = xyz[3 * p + 2];
        const int  ray = p >> 8;                        // p / NSAMP
        const float vx = viewdirs[3 * ray + 0];
        const float vy = viewdirs[3 * ray + 1];
        const float vz = viewdirs[3 * ray + 2];

        // trilinear setup (align_corners=true semantics)
        const float xs = (px + 1.f) * 0.5f * (GRD - 1);
        const float ys = (py + 1.f) * 0.5f * (GRD - 1);
        const float zs = (pz + 1.f) * 0.5f * (GRD - 1);
        int x0 = (int)floorf(xs); x0 = (x0 < 0) ? 0 : ((x0 > GRD - 2) ? GRD - 2 : x0);
        int y0 = (int)floorf(ys); y0 = (y0 < 0) ? 0 : ((y0 > GRD - 2) ? GRD - 2 : y0);
        int z0 = (int)floorf(zs); z0 = (z0 < 0) ? 0 : ((z0 > GRD - 2) ? GRD - 2 : z0);
        const float fx = xs - (float)x0, fy = ys - (float)y0, fz = zs - (float)z0;
        const int base = (z0 * GRD + y0) * GRD + x0;

        const float wx0 = 1.f - fx, wy0 = 1.f - fy, wz0 = 1.f - fz;
        const float w0 = wz0 * wy0 * wx0, w1 = wz0 * wy0 * fx;
        const float w2 = wz0 * fy  * wx0, w3 = wz0 * fy  * fx;
        const float w4 = fz  * wy0 * wx0, w5 = fz  * wy0 * fx;
        const float w6 = fz  * fy  * wx0, w7 = fz  * fy  * fx;

        // density -> softplus -> sigma scratch
        {
            const float* dv = density_vol + base;
            const float sf = w0 * dv[0]        + w1 * dv[1]
                           + w2 * dv[GRD]      + w3 * dv[GRD + 1]
                           + w4 * dv[G2]       + w5 * dv[G2 + 1]
                           + w6 * dv[G2 + GRD] + w7 * dv[G2 + GRD + 1];
            sigma_out[p] = softplus_f(sf);
        }

        // appearance features: 27 channels x 8 gathers (dominant memory cost)
#pragma unroll 3
        for (int c = 0; c < APPC; ++c) {
            const float* av = app_vol + (size_t)c * G3 + base;
            const float f = w0 * av[0]        + w1 * av[1]
                          + w2 * av[GRD]      + w3 * av[GRD + 1]
                          + w4 * av[G2]       + w5 * av[G2 + 1]
                          + w6 * av[G2 + GRD] + w7 * av[G2 + GRD + 1];
            Alds[tid][c] = (_Float16)f;
        }
        Alds[tid][27] = (_Float16)vx;
        Alds[tid][28] = (_Float16)vy;
        Alds[tid][29] = (_Float16)vz;

        // positional encodings: per-coordinate freqs, [sin(18), cos(18)] twice
        const float pc[3] = {px, py, pz};
        const float vc[3] = {vx, vy, vz};
#pragma unroll
        for (int c = 0; c < 3; ++c) {
#pragma unroll
            for (int j = 0; j < 6; ++j) {
                const float fs = (float)(1 << j);
                float s, co;
                __sincosf(pc[c] * fs, &s, &co);
                Alds[tid][30 + c * 6 + j] = (_Float16)s;
                Alds[tid][48 + c * 6 + j] = (_Float16)co;
                __sincosf(vc[c] * fs, &s, &co);
                Alds[tid][66 + c * 6 + j] = (_Float16)s;
                Alds[tid][84 + c * 6 + j] = (_Float16)co;
            }
        }
        for (int j = INDIM; j < FCW; ++j) Alds[tid][j] = (_Float16)0.f;
    }
    __syncthreads();

    const int lane = tid & 31;
    const int wv   = tid >> 5;
    const int ncol = lane & 15;
    const int rbase = wv * 16 + ((lane < 16) ? 0 : 8);

    // ---- layer 1: [128x128] x [128x128] via WMMA ----
    v8f acc[8];
#pragma unroll
    for (int nt = 0; nt < 8; ++nt)
#pragma unroll
        for (int r = 0; r < 8; ++r) acc[nt][r] = 0.f;

    wave_gemm_16x128(Alds, W1t, wv, lane, acc);

#pragma unroll
    for (int nt = 0; nt < 8; ++nt) {
        const int col = nt * 16 + ncol;
        const float bb = b1s[col];
#pragma unroll
        for (int r = 0; r < 8; ++r) {
            const float v = acc[nt][r] + bb;
            Hlds[rbase + r][col] = (_Float16)fmaxf(v, 0.f);
        }
    }
    __syncthreads();

    // ---- layer 2 ----
#pragma unroll
    for (int nt = 0; nt < 8; ++nt)
#pragma unroll
        for (int r = 0; r < 8; ++r) acc[nt][r] = 0.f;

    wave_gemm_16x128(Hlds, W2t, wv, lane, acc);
    __syncthreads();   // everyone done reading Hlds before it is overwritten

#pragma unroll
    for (int nt = 0; nt < 8; ++nt) {
        const int col = nt * 16 + ncol;
        const float bb = b2s[col];
#pragma unroll
        for (int r = 0; r < 8; ++r) {
            const float v = acc[nt][r] + bb;
            Hlds[rbase + r][col] = (_Float16)fmaxf(v, 0.f);
        }
    }
    __syncthreads();

    // ---- layer 3: 128 -> 3, VALU (3-col WMMA tile would waste the array) ----
    if (tid < ROWS) {
        const int p = blk * ROWS + tid;
        float o0 = b3s[0], o1 = b3s[1], o2 = b3s[2];
#pragma unroll 8
        for (int k = 0; k < FCW; ++k) {
            const float h = (float)Hlds[tid][k];
            o0 = fmaf(h, W3s[k * 3 + 0], o0);
            o1 = fmaf(h, W3s[k * 3 + 1], o1);
            o2 = fmaf(h, W3s[k * 3 + 2], o2);
        }
        rgb_out[3 * p + 0] = sigmoid_f(o0);
        rgb_out[3 * p + 1] = sigmoid_f(o1);
        rgb_out[3 * p + 2] = sigmoid_f(o2);
    }
}

// Sequential cumprod compositing: one thread per ray.
__global__ __launch_bounds__(128)
void composite_kernel(const float* __restrict__ sigma,
                      const float* __restrict__ rgb,
                      const float* __restrict__ dists,
                      float* __restrict__ out)
{
    const int ray = blockIdx.x * blockDim.x + threadIdx.x;
    if (ray >= NRAYS) return;
    float T = 1.f, r = 0.f, g = 0.f, b = 0.f;
    const int base = ray * NSAMP;
    for (int s = 0; s < NSAMP; ++s) {
        const int i = base + s;
        const float alpha = 1.f - __expf(-sigma[i] * dists[i] * 25.0f);
        const float w = alpha * T;
        r = fmaf(w, rgb[3 * i + 0], r);
        g = fmaf(w, rgb[3 * i + 1], g);
        b = fmaf(w, rgb[3 * i + 2], b);
        T *= (1.f - alpha + 1e-10f);
    }
    out[3 * ray + 0] = r;
    out[3 * ray + 1] = g;
    out[3 * ray + 2] = b;
}

extern "C" void kernel_launch(void* const* d_in, const int* in_sizes, int n_in,
                              void* d_out, int out_size, void* d_ws, size_t ws_size,
                              hipStream_t stream)
{
    const float* xyz         = (const float*)d_in[0];
    const float* viewdirs    = (const float*)d_in[1];
    const float* dists       = (const float*)d_in[2];
    const float* density_vol = (const float*)d_in[3];
    const float* app_vol     = (const float*)d_in[4];
    const float* W1          = (const float*)d_in[5];
    const float* b1          = (const float*)d_in[6];
    const float* W2          = (const float*)d_in[7];
    const float* b2          = (const float*)d_in[8];
    const float* W3          = (const float*)d_in[9];
    const float* b3          = (const float*)d_in[10];

    float* out      = (float*)d_out;
    float* sigma_ws = (float*)d_ws;            // NPTS floats
    float* rgb_ws   = sigma_ws + NPTS;         // 3*NPTS floats

    nerf_feat_mlp_kernel<<<NPTS / ROWS, THREADS, 0, stream>>>(
        xyz, viewdirs, density_vol, app_vol,
        W1, b1, W2, b2, W3, b3, sigma_ws, rgb_ws);

    composite_kernel<<<NRAYS / 128, 128, 0, stream>>>(sigma_ws, rgb_ws, dists, out);
}
